// PointFlowModuleWithMaxAvgpool_704374636883
// MI455X (gfx1250) — matrix operand order repaired and checked by reading, not compiled, not run on hardware
//
#include <hip/hip_runtime.h>
#include <stdint.h>

typedef __attribute__((ext_vector_type(16))) _Float16 v16h;
typedef __attribute__((ext_vector_type(8)))  _Float16 v8h;
typedef __attribute__((ext_vector_type(4)))  _Float16 v4h;
typedef __attribute__((ext_vector_type(8)))  float    v8f;
typedef __attribute__((ext_vector_type(4)))  int      i32x4_t;

#define DEVFN static __device__ __forceinline__

constexpr int NB  = 8, CIN = 256, DIM = 64;
constexpr int HH  = 64, WH = 64, PH = HH*WH;     // 4096
constexpr int HL  = 128, WL = 128, PL = HL*WL;   // 16384
constexpr int NPTS = 64;
constexpr int PADW = 66;                         // 64 + 1-pixel zero border
constexpr int EDGE_OFF = NB*CIN*PL;              // 33554432 floats (final), then edge_pred

// Per-lane K index inside a 16x32 f16 A-fragment (CDNA5 ISA 7.12.2 table)
DEVFN int kfragA(int l, int e){
  return ((e>>3)<<4) + ((l>>4)<<3) + (((e>>1)&3)<<1) + (e&1);
}

DEVFN v8f wmma_f32_16x16x32(v16h a, v16h b, v8f c){
  return __builtin_amdgcn_wmma_f32_16x16x32_f16(false, a, false, b, (short)0, c, false, false);
}

// ---- CDNA5 async global->LDS copy (ASYNCcnt path), guarded w/ sync fallback ----
// Probe-learned signature: (int4 addrspace(1)*, int4 addrspace(3)*, imm offset, imm cpol)
#if __has_builtin(__builtin_amdgcn_global_load_async_to_lds_b128)
#define HAVE_ASYNC_LDS 1
DEVFN void async_copy_b128(const _Float16* g, _Float16* l){
  typedef __attribute__((address_space(1))) i32x4_t* gp_t;
  typedef __attribute__((address_space(3))) i32x4_t* lp_t;
  __builtin_amdgcn_global_load_async_to_lds_b128(
      (gp_t)(uintptr_t)g, (lp_t)(uint32_t)(uintptr_t)l, 0, 0);
}
#else
#define HAVE_ASYNC_LDS 0
#endif

DEVFN void wait_asynccnt0(){
#if __has_builtin(__builtin_amdgcn_s_wait_asynccnt)
  __builtin_amdgcn_s_wait_asynccnt(0);
#else
  asm volatile("s_wait_asynccnt 0x0" ::: "memory");
#endif
}

// ---------------------------------------------------------------------------
// Weight pre-packing into WMMA A-fragment order: [mt][s][kc][lane32][e16]
// ---------------------------------------------------------------------------
__global__ __launch_bounds__(256) void k_pack_ef1w(const float* __restrict__ w, _Float16* __restrict__ wp){
  int idx = blockIdx.x*256 + threadIdx.x;           // 16*9*8*512 = 589824
  if(idx >= 16*9*8*512) return;
  int e = idx & 15; int t = idx >> 4;
  int l = t & 31; t >>= 5;
  int kc = t & 7; t >>= 3;
  int s = t % 9; int mt = t / 9;
  int m = mt*16 + (l & 15);
  int k = kc*32 + ((l >> 4) << 4) + e;
  wp[idx] = (_Float16)w[((size_t)m*256 + k)*9 + s];     // OIHW, s = ky*3+kx
}

__global__ __launch_bounds__(256) void k_pack_lepew(const float* __restrict__ w, _Float16* __restrict__ wp){
  int idx = blockIdx.x*256 + threadIdx.x;           // 16*3*8*512 = 196608
  if(idx >= 16*3*8*512) return;
  int e = idx & 15; int t = idx >> 4;
  int l = t & 31; t >>= 5;
  int kc = t & 7; t >>= 3;
  int s = t % 3; int mt = t / 3;
  int m = mt*16 + (l & 15);
  int k = kc*32 + ((l >> 4) << 4) + e;
  wp[idx] = (_Float16)w[((size_t)m*256 + k)*9 + 3 + s]; // ky=1 row only contributes
}

__global__ __launch_bounds__(256) void k_pack_w1x1(const float* __restrict__ w, _Float16* __restrict__ wp){
  int idx = blockIdx.x*256 + threadIdx.x;           // 4*8*512 = 16384
  if(idx >= 4*8*512) return;
  int e = idx & 15; int t = idx >> 4;
  int l = t & 31; t >>= 5;
  int kc = t & 7; int mt = t >> 3;
  int m = mt*16 + (l & 15);
  int k = kc*32 + ((l >> 4) << 4) + e;
  wp[idx] = (_Float16)w[(size_t)m*256 + k];
}

__global__ __launch_bounds__(256) void k_bnprep(const float* g, const float* b, const float* m,
                                                const float* v, float* s, float* t){
  int i = threadIdx.x;
  float sc = g[i] * rsqrtf(v[i] + 1e-5f);
  s[i] = sc; t[i] = b[i] - m[i]*sc;
}

// ---------------------------------------------------------------------------
// 1x1 conv (down_h / down_l) as WMMA GEMM: M=64 outch, K=256, N=pixels
// Block: 256 thr = 8 waves = 4(mt) x 2(nt), tile 64 outch x 32 pixels
// ---------------------------------------------------------------------------
__global__ __launch_bounds__(256) void k_conv1x1(const float* __restrict__ x, const _Float16* __restrict__ wp,
                                                 const float* __restrict__ bias, float* __restrict__ out, int HWp){
  int blk = blockIdx.x;
  int nblk = HWp >> 5;
  int b  = blk / nblk;
  int p0 = (blk % nblk) << 5;
  int tid = threadIdx.x, wave = tid >> 5, lane = tid & 31;
  int mt = wave >> 1, ntl = wave & 1;
  __shared__ __align__(32) _Float16 slab[32*32];     // [pix32][ch32]
  v8f acc = {};
  const float* xb = x + (size_t)b*CIN*HWp;
  for(int kc = 0; kc < 8; ++kc){
    __syncthreads();
    {
      int ch = tid >> 3, px4 = (tid & 7) << 2;       // 256 units exactly
      const float4 f = *(const float4*)(xb + (size_t)(kc*32 + ch)*HWp + p0 + px4);
      slab[(px4+0)*32 + ch] = (_Float16)f.x;
      slab[(px4+1)*32 + ch] = (_Float16)f.y;
      slab[(px4+2)*32 + ch] = (_Float16)f.z;
      slab[(px4+3)*32 + ch] = (_Float16)f.w;
    }
    __syncthreads();
    v16h a  = *(const v16h*)(wp + ((size_t)(mt*8 + kc))*512 + lane*16);
    v16h bm = *(const v16h*)(slab + ((ntl*16 + (lane & 15))*32 + ((lane >> 4) << 4)));
    acc = wmma_f32_16x16x32(a, bm, acc);
  }
  int n = p0 + ntl*16 + (lane & 15);
  int mhi = (lane >> 4) << 3;
  float* ob = out + (size_t)b*DIM*HWp;
  #pragma unroll
  for(int e = 0; e < 8; ++e){
    int o = mt*16 + mhi + e;
    ob[(size_t)o*HWp + n] = acc[e] + bias[o];
  }
}

// ---------------------------------------------------------------------------
// Bilinear resize (align-corners linspace) 128x128 -> 64x64 per (b,c) plane
// ---------------------------------------------------------------------------
__global__ __launch_bounds__(256) void k_resize_128to64(const float* __restrict__ in, float* __restrict__ out){
  int idx = blockIdx.x*256 + threadIdx.x;           // NB*DIM*PH = 2097152
  if(idx >= NB*DIM*PH) return;
  int x = idx & 63, y = (idx >> 6) & 63, bc = idx >> 12;
  float fy = (float)y*(127.0f/63.0f), fx = (float)x*(127.0f/63.0f);
  int y0 = (int)floorf(fy), x0 = (int)floorf(fx);
  int y1 = min(y0+1, 127), x1 = min(x0+1, 127);
  float wy = fy - (float)y0, wx = fx - (float)x0;
  const float* ip = in + (size_t)bc*PL;
  float r0 = ip[y0*128 + x0]*(1.f-wx) + ip[y0*128 + x1]*wx;
  float r1 = ip[y1*128 + x0]*(1.f-wx) + ip[y1*128 + x1]*wx;
  out[idx] = r0*(1.f-wy) + r1*wy;
}

// match conv3x3 (128ch -> 1) + sigmoid
__global__ __launch_bounds__(256) void k_match(const float* __restrict__ xh_e, const float* __restrict__ xl_up,
                                               const float* __restrict__ mw, const float* __restrict__ mb,
                                               float* __restrict__ cert){
  __shared__ float w[1152];
  int tid = threadIdx.x;
  for(int i = tid; i < 1152; i += 256) w[i] = mw[i];
  __syncthreads();
  int idx = blockIdx.x*256 + tid;                   // 32768
  int b = idx >> 12, p = idx & 4095, y = p >> 6, x = p & 63;
  float s = mb[0];
  const float* ph = xh_e + (size_t)b*DIM*PH;
  const float* pl = xl_up + (size_t)b*DIM*PH;
  for(int c = 0; c < 128; ++c){
    const float* src = (c < 64) ? ph + (size_t)c*PH : pl + (size_t)(c-64)*PH;
    const float* wr = w + c*9;
    #pragma unroll
    for(int dy = 0; dy < 3; ++dy){
      int yy = y + dy - 1; if((unsigned)yy >= 64u) continue;
      #pragma unroll
      for(int dx = 0; dx < 3; ++dx){
        int xx = x + dx - 1; if((unsigned)xx >= 64u) continue;
        s += wr[dy*3 + dx]*src[yy*64 + xx];
      }
    }
  }
  cert[idx] = 1.0f/(1.0f + __expf(-s));
}

__global__ __launch_bounds__(256) void k_avgpool(const float* __restrict__ cert, float* __restrict__ ag){
  int idx = blockIdx.x*256 + threadIdx.x;
  if(idx >= NB*64) return;
  int b = idx >> 6, i = (idx >> 3) & 7, j = idx & 7;
  const float* cp = cert + b*4096;
  float s = 0.f;
  for(int r = 0; r < 8; ++r)
    for(int c = 0; c < 8; ++c) s += cp[(i*8 + r)*64 + j*8 + c];
  ag[idx] = s*(1.0f/64.0f);
}

__global__ __launch_bounds__(256) void k_agup(const float* __restrict__ ag, float* __restrict__ agu){
  int idx = blockIdx.x*256 + threadIdx.x;           // 32768
  int b = idx >> 12, p = idx & 4095, y = p >> 6, x = p & 63;
  float fy = (float)y*(7.0f/63.0f), fx = (float)x*(7.0f/63.0f);
  int y0 = (int)floorf(fy), x0 = (int)floorf(fx);
  int y1 = min(y0+1, 7), x1 = min(x0+1, 7);
  float wy = fy - (float)y0, wx = fx - (float)x0;
  const float* a = ag + b*64;
  float r0 = a[y0*8 + x0]*(1.f-wx) + a[y0*8 + x1]*wx;
  float r1 = a[y1*8 + x0]*(1.f-wx) + a[y1*8 + x1]*wx;
  agu[idx] = r0*(1.f-wy) + r1*wy;
}

// zero the 1-pixel border of the padded NHWC f16 activation buffer
__global__ __launch_bounds__(256) void k_zeropad(_Float16* __restrict__ o){
  int idx = blockIdx.x*256 + threadIdx.x;           // NB*260*32 = 66560
  if(idx >= NB*260*32) return;
  int b = idx / (260*32);
  int r = idx % (260*32);
  int p = r >> 5, seg = r & 31;
  int py, px;
  if(p < 66){ py = 0; px = p; }
  else if(p < 132){ py = 65; px = p - 66; }
  else { int q = p - 132; py = 1 + (q >> 1); px = (q & 1)*65; }
  v8h z = {};
  *(v8h*)(o + (((size_t)b*PADW + py)*PADW + px)*256 + seg*8) = z;
}

// x_high_edge = x_high*(1-ag) -> f16 padded NHWC [b][66][66][256] via LDS transpose
__global__ __launch_bounds__(256) void k_xhe16(const float* __restrict__ xh, const float* __restrict__ agu,
                                               _Float16* __restrict__ o){
  int blk = blockIdx.x;                             // 8b * 64y * 4cc = 2048
  int cc = blk & 3; int y = (blk >> 2) & 63; int b = blk >> 8;
  int c0 = cc*64;
  int tid = threadIdx.x;
  __shared__ __align__(32) _Float16 tile[64*64];    // [px][ch]
  __shared__ float agrow[64];
  if(tid < 64) agrow[tid] = 1.0f - agu[(b << 12) + y*64 + tid];
  __syncthreads();
  const float* src = xh + ((size_t)b*CIN + c0)*PH + y*64;
  #pragma unroll
  for(int i = 0; i < 16; ++i){
    int e = i*256 + tid;                            // [0,4096)
    int px = e & 63, ch = e >> 6;
    tile[px*64 + ch] = (_Float16)(src[(size_t)ch*PH + px] * agrow[px]);
  }
  __syncthreads();
  _Float16* dst = o + (((size_t)b*PADW + y + 1)*PADW + 1)*256 + c0;
  for(int u = tid; u < 512; u += 256){
    int px = u >> 3, seg = u & 7;
    *(v8h*)(dst + (size_t)px*256 + seg*8) = *(const v8h*)(tile + px*64 + seg*8);
  }
}

// ---------------------------------------------------------------------------
// ef1 3x3 conv as implicit WMMA GEMM + BN + ReLU, output NHWC f16.
// Block: 256 thr = 8 waves = 4(mt local) x 2(nt); tile 64 outch x 32 pixels (one row).
// LDS slab per K-chunk: [3 rows][34 cols][32 ch] halves; source is padded NHWC,
// so each staged unit is a contiguous 16B copy -> async global->LDS when available.
// ---------------------------------------------------------------------------
__global__ __launch_bounds__(256) void k_ef1(const _Float16* __restrict__ xin, const _Float16* __restrict__ wp,
                                             const float* __restrict__ bns, const float* __restrict__ bnt,
                                             _Float16* __restrict__ hout){
  int blk = blockIdx.x;                             // 8b * 64y * 2xb * 4mb = 4096
  int mb = blk & 3; int xb = (blk >> 2) & 1; int y = (blk >> 3) & 63; int b = blk >> 9;
  int x0 = xb*32;
  int tid = threadIdx.x, wave = tid >> 5, lane = tid & 31;
  int mtl = wave >> 1, ntl = wave & 1;
  int mt = mb*4 + mtl;
  __shared__ __align__(32) _Float16 slab[3*34*32];
  v8f acc = {};
  const _Float16* xbp = xin + (((size_t)b*PADW + y)*PADW + x0)*256;  // padded coords: row y+r, col x0+c
  for(int kc = 0; kc < 8; ++kc){
    __syncthreads();
    for(int i = tid; i < 408; i += 256){            // 102 (r,c) pairs * 4 x 16B segments
      int seg = i & 3; int rc = i >> 2;
      int r = rc / 34, c = rc % 34;
      const _Float16* g = xbp + ((size_t)r*PADW + c)*256 + kc*32 + seg*8;
      _Float16* l = slab + rc*32 + seg*8;
#if HAVE_ASYNC_LDS
      async_copy_b128(g, l);
#else
      *(v8h*)l = *(const v8h*)g;
#endif
    }
    if(kc < 7) __builtin_prefetch(xbp + (kc + 1)*32, 0, 1);
#if HAVE_ASYNC_LDS
    wait_asynccnt0();
#endif
    __syncthreads();
    #pragma unroll
    for(int s = 0; s < 9; ++s){
      int dy = s/3, dx = s%3;
      v16h a  = *(const v16h*)(wp + ((size_t)((mt*9 + s)*8 + kc))*512 + lane*16);
      v16h bm = *(const v16h*)(slab + ((dy*34 + ntl*16 + (lane & 15) + dx)*32 + ((lane >> 4) << 4)));
      acc = wmma_f32_16x16x32(a, bm, acc);
    }
  }
  int p = y*64 + x0 + ntl*16 + (lane & 15);
  int obase = mt*16 + ((lane >> 4) << 3);
  float4 s0 = *(const float4*)(bns + obase), s1 = *(const float4*)(bns + obase + 4);
  float4 t0 = *(const float4*)(bnt + obase), t1 = *(const float4*)(bnt + obase + 4);
  float sv[8] = {s0.x, s0.y, s0.z, s0.w, s1.x, s1.y, s1.z, s1.w};
  float tv[8] = {t0.x, t0.y, t0.z, t0.w, t1.x, t1.y, t1.z, t1.w};
  v8h hv;
  #pragma unroll
  for(int e = 0; e < 8; ++e) hv[e] = (_Float16)fmaxf(acc[e]*sv[e] + tv[e], 0.f);
  *(v8h*)(hout + ((size_t)b*PH + p)*256 + obase) = hv;
}

// ef2: 3x3 conv 256ch -> 1 from NHWC f16 h, thread per output pixel
__global__ __launch_bounds__(256) void k_ef2(const _Float16* __restrict__ h, const float* __restrict__ w2,
                                             float* __restrict__ ep){
  __shared__ float w[2304];
  int tid = threadIdx.x;
  for(int i = tid; i < 2304; i += 256) w[i] = w2[i];
  __syncthreads();
  int idx = blockIdx.x*256 + tid;                   // 32768
  int b = idx >> 12, p = idx & 4095, y = p >> 6, x = p & 63;
  float s = 0.f;
  for(int ky = 0; ky < 3; ++ky){
    int yy = y + ky - 1; if((unsigned)yy >= 64u) continue;
    for(int kx = 0; kx < 3; ++kx){
      int xx = x + kx - 1; if((unsigned)xx >= 64u) continue;
      const _Float16* hp = h + ((size_t)b*PH + yy*64 + xx)*256;
      const float* wr = w + (ky*3 + kx);
      for(int c0 = 0; c0 < 256; c0 += 8){
        v8h hv = *(const v8h*)(hp + c0);
        #pragma unroll
        for(int j = 0; j < 8; ++j) s += (float)hv[j]*wr[(c0 + j)*9];
      }
    }
  }
  ep[idx] = s;
}

// top-64 of 4096 per batch; iterative argmax, tie -> lowest index (matches lax.top_k)
__global__ __launch_bounds__(256) void k_topk(const float* __restrict__ ep, int* __restrict__ pidx){
  __shared__ float vals[4096];
  __shared__ float rv[256];
  __shared__ int   ri[256];
  int b = blockIdx.x, tid = threadIdx.x;
  for(int i = tid; i < 4096; i += 256) vals[i] = ep[b*4096 + i];
  __syncthreads();
  for(int it = 0; it < 64; ++it){
    float bv = -__builtin_inff(); int bi = 0x7fffffff;
    for(int i = tid*16; i < tid*16 + 16; ++i){
      float v = vals[i];
      if(v > bv){ bv = v; bi = i; }
    }
    rv[tid] = bv; ri[tid] = bi;
    __syncthreads();
    for(int off = 128; off > 0; off >>= 1){
      if(tid < off){
        float ov = rv[tid + off]; int oi = ri[tid + off];
        if(ov > rv[tid] || (ov == rv[tid] && oi < ri[tid])){ rv[tid] = ov; ri[tid] = oi; }
      }
      __syncthreads();
    }
    if(tid == 0){ pidx[b*64 + it] = ri[0]; vals[ri[0]] = -__builtin_inff(); }
    __syncthreads();
  }
}

// ---------------------------------------------------------------------------
// Sample points (grid-sample collapses: x_high exact gather, x_low 2x2 avg)
// and emit all WMMA fragment-layout tensors for the attention stage.
// ---------------------------------------------------------------------------
__global__ __launch_bounds__(256) void k_sample_pack(const float* __restrict__ xh, const float* __restrict__ xl,
                                                     const int* __restrict__ pidx, float* __restrict__ lowf,
                                                     _Float16* __restrict__ lowA, _Float16* __restrict__ highB,
                                                     _Float16* __restrict__ vB, _Float16* __restrict__ lepB){
  int b = blockIdx.x, ch = threadIdx.x;
  __shared__ int pi[64];
  if(ch < 64) pi[ch] = pidx[b*64 + ch];
  __syncthreads();
  const float* xhb = xh + ((size_t)b*CIN + ch)*PH;
  const float* xlb = xl + ((size_t)b*CIN + ch)*PL;
  for(int pt = 0; pt < 64; ++pt){
    int q = pi[pt]; int px = q & 63, py = q >> 6;
    float hv = xhb[q];
    { // highB[b][nt][kc][l][e] = high[m=ch][k=pt]
      int kc = pt >> 5, lhi = (pt >> 3) & 1, e = (((pt >> 4) & 1) << 3) + (pt & 7);
      highB[((((size_t)b*16 + (ch >> 4))*2 + kc)*32 + (ch & 15) + 16*lhi)*16 + e] = (_Float16)hv;
    }
    { // vB[b][nt][kc][l][e] = high[4c + (m>>6)][m&63]
      int c = ch >> 2; int kc = ((ch & 3) << 1) + (pt >> 5); int lhi = (pt >> 3) & 1; int e = (((pt >> 4) & 1) << 3) + (pt & 7);
      vB[((((size_t)b*4 + (c >> 4))*8 + kc)*32 + (c & 15) + 16*lhi)*16 + e] = (_Float16)hv;
    }
    float lv = 0.25f*( xlb[(2*py)*WL + 2*px]     + xlb[(2*py)*WL + 2*px + 1]
                     + xlb[(2*py + 1)*WL + 2*px] + xlb[(2*py + 1)*WL + 2*px + 1] );
    lowf[((size_t)b*256 + ch)*64 + pt] = lv;
    { // lowA[b][mt][kc][l][e] = low[n=ch][k=pt]
      int kc = pt >> 5, r = pt & 31;
      int ehi = r >> 4, lhi = (r >> 3) & 1;
      int e = (ehi << 3) + (((r >> 1) & 3) << 1) + (r & 1);
      lowA[((((size_t)b*16 + (ch >> 4))*2 + kc)*32 + (ch & 15) + 16*lhi)*16 + e] = (_Float16)lv;
    }
  }
  // lepeB[b][s][kc][nt][l][e] = (p in range) ? high[ch][p] : 0, p = q_pos + s - 1
  for(int p = -1; p <= 64; ++p){
    float v = 0.f;
    if(p >= 0 && p < 64) v = xhb[pi[p]];
    _Float16 hv = (_Float16)v;
    #pragma unroll
    for(int s = 0; s < 3; ++s){
      int q = p + 1 - s;
      if(q < 0 || q >= 64) continue;
      lepB[(((((size_t)b*3 + s)*8 + (ch >> 5))*4 + (q >> 4))*32 + (q & 15) + 16*((ch >> 4) & 1))*16 + (ch & 15)] = hv;
    }
  }
}

// S[b][n][m] = sum_pts low[n,pt]*high[m,pt]  (M=N=256, K=64)
__global__ __launch_bounds__(256) void k_gemm_S(const _Float16* __restrict__ lowA, const _Float16* __restrict__ highB,
                                                float* __restrict__ S){
  int blk = blockIdx.x;                             // 8*32
  int b = blk >> 5; int t = (blk & 31)*8 + (threadIdx.x >> 5);
  int lane = threadIdx.x & 31;
  int mt = t >> 4, nt = t & 15;
  v8f acc = {};
  #pragma unroll
  for(int kc = 0; kc < 2; ++kc){
    v16h a  = *(const v16h*)(lowA  + ((((size_t)b*16 + mt)*2 + kc)*512) + lane*16);
    v16h bm = *(const v16h*)(highB + ((((size_t)b*16 + nt)*2 + kc)*512) + lane*16);
    acc = wmma_f32_16x16x32(a, bm, acc);
  }
  int col = nt*16 + (lane & 15);
  int rb = mt*16 + ((lane >> 4) << 3);
  float* Sp = S + (size_t)b*65536;
  #pragma unroll
  for(int e = 0; e < 8; ++e) Sp[(size_t)(rb + e)*256 + col] = acc[e];
}

// row softmax over m; writes attn directly in A-fragment layout for next GEMM
__global__ __launch_bounds__(256) void k_softmax(const float* __restrict__ S, _Float16* __restrict__ attnA){
  __shared__ float rb[8][256];
  int wave = threadIdx.x >> 5, lane = threadIdx.x & 31;
  int row = blockIdx.x*8 + wave;                    // 2048 rows
  int b = row >> 8, n = row & 255;
  const float* Sp = S + (size_t)row*256;
  float v[8]; float mx = -__builtin_inff();
  #pragma unroll
  for(int j = 0; j < 8; ++j){ v[j] = Sp[lane*8 + j]; mx = fmaxf(mx, v[j]); }
  #pragma unroll
  for(int o = 16; o > 0; o >>= 1) mx = fmaxf(mx, __shfl_xor(mx, o, 32));
  float sum = 0.f;
  #pragma unroll
  for(int j = 0; j < 8; ++j){ v[j] = __expf(v[j] - mx); sum += v[j]; }
  #pragma unroll
  for(int o = 16; o > 0; o >>= 1) sum += __shfl_xor(sum, o, 32);
  float inv = 1.0f/sum;
  #pragma unroll
  for(int j = 0; j < 8; ++j) rb[wave][lane*8 + j] = v[j]*inv;
  __syncthreads();
  _Float16* base = attnA + (((size_t)b*16 + (n >> 4))*8)*512;
  #pragma unroll
  for(int t = 0; t < 8; ++t){
    int u = lane*8 + t;
    int kc = u >> 5, lhi = (u >> 4) & 1, e = u & 15;
    int l = (n & 15) + 16*lhi;
    int m = kc*32 + kfragA(l, e);
    base[(size_t)kc*512 + l*16 + e] = (_Float16)rb[wave][m];
  }
}

// attnV[b][n][c] = sum_m attn[n,m]*v[m,c]  (M=256, N=64, K=256)
__global__ __launch_bounds__(256) void k_gemm_AV(const _Float16* __restrict__ attnA, const _Float16* __restrict__ vB,
                                                 float* __restrict__ attnV){
  int blk = blockIdx.x;                             // 8*8
  int b = blk >> 3; int t = (blk & 7)*8 + (threadIdx.x >> 5);
  int lane = threadIdx.x & 31;
  int mt = t >> 2, nt = t & 3;
  v8f acc = {};
  #pragma unroll
  for(int kc = 0; kc < 8; ++kc){
    v16h a  = *(const v16h*)(attnA + (((size_t)b*16 + mt)*8 + kc)*512 + lane*16);
    v16h bm = *(const v16h*)(vB    + (((size_t)b*4  + nt)*8 + kc)*512 + lane*16);
    acc = wmma_f32_16x16x32(a, bm, acc);
  }
  int c = nt*16 + (lane & 15);
  int rb = mt*16 + ((lane >> 4) << 3);
  float* op = attnV + (size_t)b*256*64;
  #pragma unroll
  for(int e = 0; e < 8; ++e) op[(size_t)(rb + e)*64 + c] = acc[e];
}

// lepeC[b][o][p] = lepe_b[o] + sum_s sum_k w[o][k][1][s]*hf[k][p+s-1]
__global__ __launch_bounds__(256) void k_gemm_lepe(const _Float16* __restrict__ wp, const _Float16* __restrict__ lepB,
                                                   const float* __restrict__ bias, float* __restrict__ outc){
  int blk = blockIdx.x;                             // 8*8
  int b = blk >> 3; int t = (blk & 7)*8 + (threadIdx.x >> 5);
  int lane = threadIdx.x & 31;
  int mt = t >> 2, nt = t & 3;
  v8f acc = {};
  for(int s = 0; s < 3; ++s){
    #pragma unroll
    for(int kc = 0; kc < 8; ++kc){
      v16h a  = *(const v16h*)(wp   + ((size_t)(mt*3 + s)*8 + kc)*512 + lane*16);
      v16h bm = *(const v16h*)(lepB + ((((size_t)b*3 + s)*8 + kc)*4 + nt)*512 + lane*16);
      acc = wmma_f32_16x16x32(a, bm, acc);
    }
  }
  int c = nt*16 + (lane & 15);
  int rb = mt*16 + ((lane >> 4) << 3);
  float* op = outc + (size_t)b*256*64;
  #pragma unroll
  for(int e = 0; e < 8; ++e) op[(size_t)(rb + e)*64 + c] = acc[e] + bias[rb + e];
}

__global__ __launch_bounds__(256) void k_copy(const float4* __restrict__ in, float4* __restrict__ out){
  int idx = blockIdx.x*256 + threadIdx.x;           // 8388608
  out[idx] = in[idx];
}

// fusion = out_a(+lepe, transposed-reshape) + low_feat; scatter into final
__global__ __launch_bounds__(256) void k_scatter(const float* __restrict__ attnV, const float* __restrict__ lepeC,
                                                 const float* __restrict__ lowf, const int* __restrict__ pidx,
                                                 float* __restrict__ out){
  int idx = blockIdx.x*256 + threadIdx.x;           // 131072
  int b = idx >> 14; int r = idx & 16383; int i = r >> 6; int j = r & 63;
  int q = i*64 + j; int c = q >> 8; int n = q & 255;
  float val = attnV[((size_t)b*256 + n)*64 + c]
            + lepeC[((size_t)b*256 + 4*c + (n >> 6))*64 + (n & 63)]
            + lowf[idx];
  int pq = pidx[b*64 + j];
  int col = 2*(pq & 63) + 256*(pq >> 6);
  out[((size_t)b*256 + i)*(size_t)PL + col] = val;
}

// ---------------------------------------------------------------------------
extern "C" void kernel_launch(void* const* d_in, const int* in_sizes, int n_in,
                              void* d_out, int out_size, void* d_ws, size_t ws_size,
                              hipStream_t stream){
  (void)in_sizes; (void)n_in; (void)out_size; (void)ws_size;
  const float* x_high   = (const float*)d_in[0];
  const float* x_low    = (const float*)d_in[1];
  const float* down_h_w = (const float*)d_in[2];
  const float* down_h_b = (const float*)d_in[3];
  const float* down_l_w = (const float*)d_in[4];
  const float* down_l_b = (const float*)d_in[5];
  const float* match_w  = (const float*)d_in[6];
  const float* match_b  = (const float*)d_in[7];
  const float* ef1_w    = (const float*)d_in[8];
  const float* bn_g     = (const float*)d_in[9];
  const float* bn_b     = (const float*)d_in[10];
  const float* bn_m     = (const float*)d_in[11];
  const float* bn_v     = (const float*)d_in[12];
  const float* ef2_w    = (const float*)d_in[13];
  const float* lepe_w   = (const float*)d_in[14];
  const float* lepe_b   = (const float*)d_in[15];

  char* ws = (char*)d_ws;
  size_t cur = 0;
  auto take = [&](size_t bytes)->char*{
    char* p = ws + cur;
    cur += (bytes + 255) & ~(size_t)255;
    return p;
  };
  float* xh_e     = (float*)take((size_t)NB*DIM*PH*4);
  float* xl_up    = (float*)take((size_t)NB*DIM*PH*4);
  float* cert     = (float*)take((size_t)NB*PH*4);
  float* agp      = (float*)take((size_t)NB*64*4);
  float* agu      = (float*)take((size_t)NB*PH*4);
  float* bns      = (float*)take(256*4);
  float* bnt      = (float*)take(256*4);
  _Float16* wpe1  = (_Float16*)take((size_t)16*9*8*512*2);
  _Float16* wplp  = (_Float16*)take((size_t)16*3*8*512*2);
  _Float16* wpdh  = (_Float16*)take((size_t)4*8*512*2);
  _Float16* wpdl  = (_Float16*)take((size_t)4*8*512*2);
  int* pidx       = (int*)take((size_t)NB*NPTS*4);
  float* lowf     = (float*)take((size_t)NB*CIN*NPTS*4);
  _Float16* lowA  = (_Float16*)take((size_t)NB*16*2*512*2);
  _Float16* highB = (_Float16*)take((size_t)NB*16*2*512*2);
  _Float16* vB    = (_Float16*)take((size_t)NB*4*8*512*2);
  _Float16* lepB  = (_Float16*)take((size_t)NB*3*8*4*512*2);
  float* S        = (float*)take((size_t)NB*256*256*4);
  _Float16* attnA = (_Float16*)take((size_t)NB*16*8*512*2);
  float* attnV    = (float*)take((size_t)NB*256*64*4);
  float* lepeC    = (float*)take((size_t)NB*256*64*4);
  // big aliased region: xl_e (f32) first, then reused as padded-NHWC xhe16 + h16
  constexpr size_t XLE_BYTES = (size_t)NB*DIM*PL*4;                 // 33,554,432
  constexpr size_t XHEP_BYTES = (size_t)NB*PADW*PADW*CIN*2;         // 17,842,176
  constexpr size_t H16_BYTES = (size_t)NB*CIN*PH*2;                 // 16,777,216
  constexpr size_t BIG = (XHEP_BYTES + H16_BYTES > XLE_BYTES) ? (XHEP_BYTES + H16_BYTES) : XLE_BYTES;
  char* big       = take(BIG);
  float*    xl_e  = (float*)big;
  _Float16* xhe16 = (_Float16*)big;
  _Float16* h16   = (_Float16*)(big + XHEP_BYTES);

  float* outF = (float*)d_out;
  float* edge = outF + EDGE_OFF;

  k_pack_ef1w <<<2304, 256, 0, stream>>>(ef1_w, wpe1);
  k_pack_lepew<<< 768, 256, 0, stream>>>(lepe_w, wplp);
  k_pack_w1x1 <<<  64, 256, 0, stream>>>(down_h_w, wpdh);
  k_pack_w1x1 <<<  64, 256, 0, stream>>>(down_l_w, wpdl);
  k_bnprep    <<<   1, 256, 0, stream>>>(bn_g, bn_b, bn_m, bn_v, bns, bnt);

  k_conv1x1<<<NB*(PH/32), 256, 0, stream>>>(x_high, wpdh, down_h_b, xh_e, PH);
  k_conv1x1<<<NB*(PL/32), 256, 0, stream>>>(x_low,  wpdl, down_l_b, xl_e, PL);
  k_resize_128to64<<<8192, 256, 0, stream>>>(xl_e, xl_up);
  k_match  <<<128, 256, 0, stream>>>(xh_e, xl_up, match_w, match_b, cert);
  k_avgpool<<<  2, 256, 0, stream>>>(cert, agp);
  k_agup   <<<128, 256, 0, stream>>>(agp, agu);
  k_zeropad<<<260, 256, 0, stream>>>(xhe16);
  k_xhe16  <<<2048, 256, 0, stream>>>(x_high, agu, xhe16);

  k_ef1 <<<4096, 256, 0, stream>>>(xhe16, wpe1, bns, bnt, h16);
  k_ef2 <<< 128, 256, 0, stream>>>(h16, ef2_w, edge);
  k_topk<<<  NB, 256, 0, stream>>>(edge, pidx);

  k_sample_pack<<<NB, 256, 0, stream>>>(x_high, x_low, pidx, lowf, lowA, highB, vB, lepB);
  k_gemm_S   <<<NB*32, 256, 0, stream>>>(lowA, highB, S);
  k_softmax  <<<  256, 256, 0, stream>>>(S, attnA);
  k_gemm_AV  <<< NB*8, 256, 0, stream>>>(attnA, vB, attnV);
  k_gemm_lepe<<< NB*8, 256, 0, stream>>>(wplp, lepB, lepe_b, lepeC);

  k_copy   <<<32768, 256, 0, stream>>>((const float4*)x_low, (float4*)outF);
  k_scatter<<<  512, 256, 0, stream>>>(attnV, lepeC, lowf, pidx, outF);
}
